// Attention_dot3_67156108640598
// MI455X (gfx1250) — matrix operand order repaired and checked by reading, not compiled, not run on hardware
//
#include <hip/hip_runtime.h>
#include <hip/hip_bf16.h>

typedef __attribute__((ext_vector_type(16))) __bf16 v16bf;
typedef __attribute__((ext_vector_type(8)))  float  v8f;
typedef __attribute__((ext_vector_type(4)))  int    v4i;

#define BATCH 8
#define CH    256
#define CKQ   64
#define NPIX  4096

#if defined(__HIP_DEVICE_COMPILE__) && __has_builtin(__builtin_amdgcn_global_load_async_to_lds_b128)
#define HAVE_ASYNC_LDS 1
#else
#define HAVE_ASYNC_LDS 0
#endif

typedef __attribute__((address_space(1))) v4i as1_v4i;   // global <4 x i32>
typedef __attribute__((address_space(3))) v4i as3_v4i;   // LDS    <4 x i32>

// Global -> LDS 16B copy. Async (ASYNCcnt-tracked, LDS written directly by the
// async path, no VGPR round-trip) when the toolchain exposes the gfx1250
// builtin; otherwise a plain vector copy.
static __device__ __forceinline__ void cp_b128(void* lds, const void* g) {
#if HAVE_ASYNC_LDS
    __builtin_amdgcn_global_load_async_to_lds_b128(
        (as1_v4i*)g, (as3_v4i*)lds, 0, 0);
#else
    *(uint4*)lds = *(const uint4*)g;
#endif
}
static __device__ __forceinline__ void cp_commit_wait() {
#if HAVE_ASYNC_LDS
    asm volatile("s_wait_asynccnt 0" ::: "memory");
#endif
}

static __device__ __forceinline__ v8f wmma_bf16(v16bf a, v16bf b, v8f c) {
    // D = A(16x32 bf16) * B(32x16 bf16) + C(16x16 f32)
    return __builtin_amdgcn_wmma_f32_16x16x32_bf16(false, a, false, b, (short)0, c, false, false);
}

union LdsVec {
    uint4 q[2];
    v16bf v;
};

// Two 16B LDS loads -> one 16-element bf16 WMMA operand.
static __device__ __forceinline__ v16bf ld_pair(const __bf16* p0, const __bf16* p1) {
    LdsVec t;
    t.q[0] = *(const uint4*)p0;
    t.q[1] = *(const uint4*)p1;
    return t.v;
}
// 32B contiguous LDS -> operand.
static __device__ __forceinline__ v16bf ld_contig(const __bf16* p) {
    return ld_pair(p, p + 8);
}

// -----------------------------------------------------------------------------
// Kernel 0a: pack Wcat = [Wq(64); Wk(64); Wv(256)] rows, f32 -> bf16 [384][256].
// -----------------------------------------------------------------------------
__global__ __launch_bounds__(256) void pack_w_kernel(
    const float* __restrict__ Wq, const float* __restrict__ Wk,
    const float* __restrict__ Wv, __bf16* __restrict__ Wcat)
{
    int i = blockIdx.x * 256 + threadIdx.x;     // grid covers 384*256
    int r = i >> 8, c = i & 255;
    float w = (r < 64)  ? Wq[r * CH + c]
            : (r < 128) ? Wk[(r - 64) * CH + c]
                        : Wv[(r - 128) * CH + c];
    Wcat[i] = (__bf16)w;
}

// -----------------------------------------------------------------------------
// Kernel 0b: x [B,C,N] f32 -> xt [B,N,C] bf16 (transpose via LDS; both global
// reads (along n) and writes (along c) coalesced / vectorized).
// -----------------------------------------------------------------------------
__global__ __launch_bounds__(256) void pack_x_kernel(
    const float* __restrict__ x, __bf16* __restrict__ xt)
{
    __shared__ __align__(16) __bf16 sT[64][264];   // [n][c], pad keeps rows 16B-aligned
    const int tid = threadIdx.x;
    const int n0  = blockIdx.x * 64;
    const int b   = blockIdx.y;

    for (int i = tid; i < 64 * 256; i += 256) {
        int c = i >> 6, nn = i & 63;               // read coalesced along n
        sT[nn][c] = (__bf16)x[((size_t)b * CH + c) * NPIX + n0 + nn];
    }
    __syncthreads();
    for (int i = tid; i < 64 * 32; i += 256) {     // 64 rows x 32 uint4 (512B/row)
        int row = i >> 5, part = i & 31;
        uint4* dst = (uint4*)(xt + ((size_t)b * NPIX + n0 + row) * CH);
        dst[part] = *(const uint4*)&sT[row][part * 8];
    }
}

// -----------------------------------------------------------------------------
// Kernel 1: fused projection GEMM.  P = Wcat(384x256) * X(256x64) + bias.
// Tile map: tile = tr*8 + wave  => tr=0: waves 0-3 Qt, waves 4-7 Kp; tr=1,2: V.
// Grid: (N/64, B), Block: 256 (8 waves).
// -----------------------------------------------------------------------------
__global__ __launch_bounds__(256) void proj_gemm_kernel(
    const __bf16* __restrict__ Wcat, const __bf16* __restrict__ xt,
    const float* __restrict__ bq, const float* __restrict__ bk,
    const float* __restrict__ bv,
    __bf16* __restrict__ Qt, __bf16* __restrict__ Kp, __bf16* __restrict__ V)
{
    __shared__ __align__(16) __bf16 sW[384][32];    // weight K-chunk    24 KB
    __shared__ __align__(16) __bf16 sXt[64][40];    // x^T K-chunk (pad)  5 KB

    const int tid    = threadIdx.x;
    const int wave   = tid >> 5;
    const int lane   = tid & 31;
    const int laneLo = lane & 15;
    const int laneHi = lane >> 4;
    const int n0     = blockIdx.x * 64;
    const int b      = blockIdx.y;

    v8f zero = {};
    v8f acc[3][4];
#pragma unroll
    for (int i = 0; i < 3; ++i)
#pragma unroll
        for (int j = 0; j < 4; ++j) acc[i][j] = zero;

    for (int k0 = 0; k0 < CH; k0 += 32) {
        // 384x32 weight chunk: 1536 x b128, 6/thread (async -> LDS direct)
        for (int i = tid; i < 1536; i += 256) {
            int r = i >> 2, p = i & 3;
            cp_b128(&sW[r][p * 8], Wcat + r * CH + k0 + p * 8);
        }
        // 64x32 activation chunk from xt: 256 x b128, 1/thread
        {
            int row = tid >> 2, p = tid & 3;
            cp_b128(&sXt[row][p * 8],
                    xt + ((size_t)b * NPIX + n0 + row) * CH + k0 + p * 8);
        }
        cp_commit_wait();
        __syncthreads();

        // B operands: lane = pixel column, k = 8h+s / 16+8h+(s-8) per lane half.
        v16bf Bop[4];
#pragma unroll
        for (int tc = 0; tc < 4; ++tc)
            Bop[tc] = ld_pair(&sXt[tc * 16 + laneLo][laneHi * 8], &sXt[tc * 16 + laneLo][16 + laneHi * 8]);

#pragma unroll
        for (int tr = 0; tr < 3; ++tr) {
            int row = (tr * 8 + wave) * 16 + laneLo;
            v16bf Aop = ld_pair(&sW[row][laneHi * 8], &sW[row][16 + laneHi * 8]);
#pragma unroll
            for (int tc = 0; tc < 4; ++tc)
                acc[tr][tc] = wmma_bf16(Aop, Bop[tc], acc[tr][tc]);
        }
        __syncthreads();
    }

    // ---- epilogue: destination uniform per (tr, wave) ----
    if (wave < 4) {
        // tr = 0, waves 0-3: Qt rows (8 consecutive rows per lane -> b128 store)
#pragma unroll
        for (int tc = 0; tc < 4; ++tc) {
            int ncol = n0 + tc * 16 + laneLo;
            int rb   = wave * 16 + laneHi * 8;
            union { __bf16 h[8]; uint4 q; } ep;
#pragma unroll
            for (int r = 0; r < 8; ++r)
                ep.h[r] = (__bf16)(acc[0][tc][r] + bq[rb + r]);
            *(uint4*)(Qt + ((size_t)b * NPIX + ncol) * CKQ + rb) = ep.q;
        }
    } else {
        // tr = 0, waves 4-7: Kp rows 0..63
#pragma unroll
        for (int tc = 0; tc < 4; ++tc) {
            int ncol = n0 + tc * 16 + laneLo;
#pragma unroll
            for (int r = 0; r < 8; ++r) {
                int row = (wave - 4) * 16 + r + 8 * laneHi;
                Kp[((size_t)b * CKQ + row) * NPIX + ncol] =
                    (__bf16)(acc[0][tc][r] + bk[row]);
            }
        }
    }
    // tr = 1,2: V rows 0..255
#pragma unroll
    for (int tr = 1; tr < 3; ++tr) {
#pragma unroll
        for (int tc = 0; tc < 4; ++tc) {
            int ncol = n0 + tc * 16 + laneLo;
#pragma unroll
            for (int r = 0; r < 8; ++r) {
                int row = (tr * 8 + wave - 8) * 16 + r + 8 * laneHi;
                V[((size_t)b * CH + row) * NPIX + ncol] =
                    (__bf16)(acc[tr][tc][r] + bv[row]);
            }
        }
    }
}

// -----------------------------------------------------------------------------
// Kernel 2: fused attention. Per WG: batch b, 64-column out tile.
// Double-buffered: chunk i+1's Qt/V loads are issued (async -> LDS) before
// computing chunk i, overlapping global traffic with 10 WMMAs/wave.
// Grid: (N/64, B), Block: 256 (8 waves).
// -----------------------------------------------------------------------------
__global__ __launch_bounds__(256) void attn_kernel(
    const __bf16* __restrict__ Qt, const __bf16* __restrict__ Kp,
    const __bf16* __restrict__ V,  const float* __restrict__ x,
    const float* __restrict__ gamma, float* __restrict__ out)
{
    __shared__ __align__(16) __bf16 sKt[64][64];      // K tile transposed [m][c]  8 KB
    __shared__ __align__(16) __bf16 sQ[2][32][64];    // Qt chunk [n][c] x2        8 KB
    __shared__ __align__(16) __bf16 sV[2][256][32];   // V chunk  [c][n] x2       32 KB
    __shared__ __align__(16) __bf16 sEt[64][32];      // sigmoid transposed [m][n] 4 KB

    const int tid    = threadIdx.x;
    const int wave   = tid >> 5;
    const int lane   = tid & 31;
    const int laneLo = lane & 15;
    const int laneHi = lane >> 4;
    const int m0     = blockIdx.x * 64;
    const int b      = blockIdx.y;

    auto load_chunk = [&](int nc2, int buf) {
        // Qt chunk: 32 rows x 128B -> 256 x b128, 1 per thread.
        int row = tid >> 3, part = tid & 7;
        cp_b128(&sQ[buf][row][part * 8],
                Qt + ((size_t)b * NPIX + nc2 + row) * CKQ + part * 8);
        // V chunk: 256 rows x 64B -> thread t copies its whole row (4 x b128).
        const __bf16* g = V + ((size_t)b * CH + tid) * NPIX + nc2;
#pragma unroll
        for (int p = 0; p < 4; ++p)
            cp_b128(&sV[buf][tid][p * 8], g + p * 8);
    };

    // Load K tile, transposing into sKt[m][c]; global reads coalesced along m.
    for (int i = tid; i < 64 * 64; i += 256) {
        int c = i >> 6, mm = i & 63;
        sKt[mm][c] = Kp[((size_t)b * CKQ + c) * NPIX + m0 + mm];
    }
    // Prologue: bring in chunk 0.
    load_chunk(0, 0);
    cp_commit_wait();
    __syncthreads();

    const int tr = wave >> 2;   // this wave's E tile (32x64 -> 2x4 tiles of 16x16)
    const int tc = wave & 3;

    // Loop-invariant K-tile B-operands (column = m, c = 8h+s / 16+8h+(s-8) per lane half).
    v16bf Bk[2];
#pragma unroll
    for (int koi = 0; koi < 2; ++koi)
        Bk[koi] = ld_pair(&sKt[tc * 16 + laneLo][koi * 32 + laneHi * 8], &sKt[tc * 16 + laneLo][koi * 32 + 16 + laneHi * 8]);

    v8f zero = {};
    v8f oacc[8];   // 2 row-tiles x 4 col-tiles of the 256x64 output
#pragma unroll
    for (int i = 0; i < 8; ++i) oacc[i] = zero;

    int cur = 0;
    for (int nc = 0; nc < NPIX; nc += 32) {
        // Issue next chunk's loads into the other buffer (overlaps the WMMAs).
        if (nc + 32 < NPIX) load_chunk(nc + 32, cur ^ 1);

        // ---- E phase: E(32x64) = sigmoid(Qchunk(32 x 64c) * Ktile(64c x 64m)) ----
        v8f eacc = zero;
#pragma unroll
        for (int koi = 0; koi < 2; ++koi) {
            const int ko = koi * 32;
            v16bf Aq = ld_pair(&sQ[cur][tr * 16 + laneLo][ko + laneHi * 8],
                               &sQ[cur][tr * 16 + laneLo][ko + 16 + laneHi * 8]);
            eacc = wmma_bf16(Aq, Bk[koi], eacc);
        }
        // sigmoid + bf16: 8 contiguous n per lane in transposed sEt -> one b128 store
        {
            union { __bf16 h[8]; uint4 q; } ep;
#pragma unroll
            for (int r = 0; r < 8; ++r)
                ep.h[r] = (__bf16)(1.0f / (1.0f + __expf(-eacc[r])));
            *(uint4*)&sEt[tc * 16 + laneLo][tr * 16 + laneHi * 8] = ep.q;
        }
        __syncthreads();

        // ---- out phase: oacc(256x64) += V(256x32) * E(32x64) ----
        v16bf Be[4];
#pragma unroll
        for (int t2 = 0; t2 < 4; ++t2)
            Be[t2] = ld_pair(&sEt[t2 * 16 + laneLo][laneHi * 8], &sEt[t2 * 16 + laneLo][16 + laneHi * 8]);
#pragma unroll
        for (int a2 = 0; a2 < 2; ++a2) {
            int crow = (wave * 2 + a2) * 16 + laneLo;
            v16bf Av = ld_pair(&sV[cur][crow][laneHi * 8],
                               &sV[cur][crow][16 + laneHi * 8]);
#pragma unroll
            for (int t2 = 0; t2 < 4; ++t2)
                oacc[a2 * 4 + t2] = wmma_bf16(Av, Be[t2], oacc[a2 * 4 + t2]);
        }

        // Next buffer must be resident (my async loads + everyone else's) and
        // everyone must be done with sEt before it is overwritten.
        cp_commit_wait();
        __syncthreads();
        cur ^= 1;
    }

    // ---- epilogue: out = clip(gamma)/N^2 * oacc + x ----
    float g = gamma[0];
    g = fminf(fmaxf(g, -1.0f), 1.0f);
    const float scale = g / ((float)NPIX * (float)NPIX);
#pragma unroll
    for (int a2 = 0; a2 < 2; ++a2) {
#pragma unroll
        for (int t2 = 0; t2 < 4; ++t2) {
#pragma unroll
            for (int r = 0; r < 8; ++r) {
                int c = (wave * 2 + a2) * 16 + r + 8 * laneHi;
                int m = m0 + t2 * 16 + laneLo;
                size_t idx = ((size_t)b * CH + c) * NPIX + m;
                out[idx] = scale * oacc[a2 * 4 + t2][r] + x[idx];
            }
        }
    }
}

// -----------------------------------------------------------------------------
extern "C" void kernel_launch(void* const* d_in, const int* in_sizes, int n_in,
                              void* d_out, int out_size, void* d_ws, size_t ws_size,
                              hipStream_t stream) {
    const float* x     = (const float*)d_in[0];
    const float* Wq    = (const float*)d_in[1];
    const float* bq    = (const float*)d_in[2];
    const float* Wk    = (const float*)d_in[3];
    const float* bk    = (const float*)d_in[4];
    const float* Wv    = (const float*)d_in[5];
    const float* bv    = (const float*)d_in[6];
    const float* gamma = (const float*)d_in[7];
    float* out = (float*)d_out;

    // Workspace (bf16): Qt [B,N,64] 4MB | Kp [B,64,N] 4MB | V [B,256,N] 16MB |
    //                   xt [B,N,256] 16MB | Wcat [384][256] 192KB
    char* ws = (char*)d_ws;
    __bf16* Qt   = (__bf16*)(ws);
    __bf16* Kp   = (__bf16*)(ws + (size_t)4  * 1024 * 1024);
    __bf16* V    = (__bf16*)(ws + (size_t)8  * 1024 * 1024);
    __bf16* xt   = (__bf16*)(ws + (size_t)24 * 1024 * 1024);
    __bf16* Wcat = (__bf16*)(ws + (size_t)40 * 1024 * 1024);

    dim3 block(256);
    pack_w_kernel<<<dim3(384 * 256 / 256), block, 0, stream>>>(Wq, Wk, Wv, Wcat);
    pack_x_kernel<<<dim3(NPIX / 64, BATCH), block, 0, stream>>>(x, xt);
    proj_gemm_kernel<<<dim3(NPIX / 64, BATCH), block, 0, stream>>>(
        Wcat, xt, bq, bk, bv, Qt, Kp, V);
    attn_kernel<<<dim3(NPIX / 64, BATCH), block, 0, stream>>>(Qt, Kp, V, x, gamma, out);
}
